// Decoder_71159018160511
// MI455X (gfx1250) — compile-verified, matrix operand order checked by reading, not run on hardware
//
#include <hip/hip_runtime.h>
#include <stdint.h>

// ---------------- problem dims ----------------
#define NWG   16
#define NTHR  256
#define Bdim  32
#define Tdim  256
#define Ldim  512
#define Edim  512
#define Ddim  80
#define Adim  128
#define Cdim  512
#define KMIX  5
#define EA    640   // E + A
constexpr size_t BLD = (size_t)Bdim * Ldim * Ddim;

// ---------------- WMMA / TDM types ----------------
typedef __attribute__((ext_vector_type(16))) __bf16 v16bf;
typedef __attribute__((ext_vector_type(8)))  float  v8f;
typedef __attribute__((ext_vector_type(4)))  uint32_t v4u;
typedef __attribute__((ext_vector_type(8)))  int    v8i;
typedef __attribute__((ext_vector_type(4)))  int    v4i;
union FragU { v16bf v; uint32_t u[8]; };

__device__ inline uint32_t f2bf_u(float f) {
  union { float f; uint32_t u; } x; x.f = f;
  return ((x.u + 0x7fffu + ((x.u >> 16) & 1u)) >> 16) & 0xffffu;
}
__device__ inline float sigm(float x) { return 1.0f / (1.0f + __expf(-x)); }
__device__ inline float softplusf(float x) { return (x > 20.f) ? x : __logf(1.0f + __expf(x)); }

// ---------------- workspace layout (bytes) ----------------
constexpr size_t OFF_BAR   = 0;                          // barrier counter (256B)
constexpr size_t OFF_AFTER = 256;                        // B*D f32 carry
constexpr size_t OFF_CTX   = OFF_AFTER + 10240;          // B*E bf16 carry/exchange
constexpr size_t OFF_STATE = OFF_CTX   + 32768;          // B*A f32 exchange
constexpr size_t OFF_G1    = OFF_STATE + 16384;          // B*240 f32
constexpr size_t OFF_G2    = OFF_G1    + 30720;          // B*240 f32
constexpr size_t OFF_XP0   = OFF_G2    + 30720;          // B*C bf16 ping
constexpr size_t OFF_XP1   = OFF_XP0   + 32768;          // B*C bf16 pong
constexpr size_t OFF_WPRE1 = OFF_XP1   + 32768;          // 15 frags
constexpr size_t OFF_WPRE2 = OFF_WPRE1 + 15360;
constexpr size_t OFF_WD    = OFF_WPRE2 + 15360;          // 24 frags
constexpr size_t OFF_WC    = OFF_WD    + 24576;          // 128 frags
constexpr size_t OFF_WIH1  = OFF_WC    + 131072;         // 400 frags
constexpr size_t OFF_WIH2  = OFF_WIH1  + 409600;         // 60 frags
constexpr size_t OFF_C0    = OFF_WIH2  + 61440;          // 96 frags
constexpr size_t OFF_C1    = OFF_C0    + 98304;          // 512 frags
constexpr size_t OFF_C2    = OFF_C1    + 524288;
constexpr size_t OFF_C3    = OFF_C2    + 524288;
constexpr size_t OFF_C4    = OFF_C3    + 524288;         // 80 frags

// ---------------- LDS layout (bytes) ----------------
constexpr int L_WC    = 0;                  // 16 frags (wg<8)
constexpr int L_WIH1  = L_WC    + 16*1024;  // 20 frags (wg<15)
constexpr int L_WIH2  = L_WIH1  + 20*1024;  // 3 frags
constexpr int L_C0    = L_WIH2  + 3*1024;   // 6 frags
constexpr int L_C123  = L_C0    + 6*1024;   // 96 frags
constexpr int L_C4    = L_C123  + 96*1024;  // 16 frags (wg<5)
constexpr int L_STATE = L_C4    + 16*1024;  // B*A f32
constexpr int L_MU    = L_STATE + Bdim*Adim*4;
constexpr int L_WMIX  = L_MU    + Bdim*KMIX*4;
constexpr int L_SMIX  = L_WMIX  + Bdim*KMIX*4;
constexpr int L_PA    = L_SMIX  + Bdim*KMIX*4;  // bf16 32x96 (afterA / pre2-out / h1A)
constexpr int L_PB    = L_PA    + Bdim*96*2;    // bf16 32x96 (pre1-out)
constexpr int L_REUSE = L_PB    + Bdim*96*2;    // a(f32 32x256) then rnnA(bf16 32x640)
constexpr int L_XA    = L_REUSE + Bdim*EA*2;    // bf16 32x512 (ctxA / x act)
constexpr int L_BEFA  = L_XA    + Bdim*Cdim*2;  // bf16 32x96
constexpr int L_BEFF  = L_BEFA  + Bdim*96*2;    // f32 32x80
constexpr int L_RED   = L_BEFF  + Bdim*Ddim*4;  // f32 reduction (<=32x32)
constexpr int SMEM_BYTES = L_RED + 4096;        // ~285KB < 320KB/WGP

// ---------------- Tensor Data Mover: 2-D tile Global -> LDS ----------------
// D# per CDNA5 ISA: group0 = {flags, lds_addr, gaddr, type=2},
// group1 = {data_size=2B, tensor_dim0/1, tile_dim0/1, tensor_dim0_stride}; groups 2/3 zero (<=2D).
// clang-23 toolchain: 6-arg builtin (g0 u32x4, g1 i32x8, g2 i32x4, g3 i32x4, i32x8, cpol).
__device__ inline void tdm_load_tile(uint32_t ldsOff, const void* gptr,
                                     uint32_t eprRow, uint32_t rows, uint32_t strideElems) {
  uint64_t ga = (uint64_t)(uintptr_t)gptr;
  v4u g0;
  g0[0] = 1u;                                               // count=1
  g0[1] = ldsOff;                                           // lds_addr (bytes)
  g0[2] = (uint32_t)ga;                                     // global_addr[31:0]
  g0[3] = (uint32_t)((ga >> 32) & 0x01FFFFFFu) | (2u << 30);// addr[56:32] | type=2
  v8i g1;
  g1[0] = (int)(1u << 16);                                  // workgroup_mask=0, data_size=1 (2B)
  g1[1] = (int)((eprRow & 0xFFFFu) << 16);                  // tensor_dim0[15:0]
  g1[2] = (int)(((eprRow >> 16) & 0xFFFFu) | ((rows & 0xFFFFu) << 16));   // td0[31:16] | td1[15:0]
  g1[3] = (int)(((rows >> 16) & 0xFFFFu) | ((eprRow & 0xFFFFu) << 16));   // td1[31:16] | tile_dim0
  g1[4] = (int)(rows & 0xFFFFu);                            // tile_dim1 (tile_dim2=0)
  g1[5] = (int)strideElems;                                 // tensor_dim0_stride[31:0]
  g1[6] = 0;
  g1[7] = 0;
  v4i z4; z4[0] = z4[1] = z4[2] = z4[3] = 0;
  v8i z8;
#pragma unroll
  for (int i = 0; i < 8; ++i) z8[i] = 0;
  __builtin_amdgcn_tensor_load_to_lds(g0, g1, z4, z4, z8, 0);
}
__device__ inline void tdm_wait() { __builtin_amdgcn_s_wait_tensorcnt(0); }

// ---------------- WMMA helpers ----------------
__device__ inline v8f wmma_acc(const uint32_t* a8, const uint32_t* b8, v8f c) {
  FragU A, Bf;
#pragma unroll
  for (int i = 0; i < 8; ++i) { A.u[i] = a8[i]; Bf.u[i] = b8[i]; }
  return __builtin_amdgcn_wmma_f32_16x16x32_bf16(false, A.v, false, Bf.v, (short)0, c, false, false);
}
// A frag per ISA 16-bit 16x32 layout, from bf16 row-major buffer (stride S elems)
__device__ inline void load_afrag(const uint16_t* Ab, int S, int mt, int kt, uint32_t* u) {
  int lane = threadIdx.x & 31;
  int row  = mt * 16 + (lane & 15);
  int half = lane >> 4;
  const uint16_t* rp = Ab + row * S + kt * 32;
#pragma unroll
  for (int v = 0; v < 8; ++v) {
    int k = (v < 4) ? (8 * half + 2 * v) : (16 + 8 * half + 2 * (v - 4));
    u[v] = *(const uint32_t*)(rp + k);
  }
}
// B frag: pre-swizzled, 8 contiguous dwords per lane
__device__ inline void load_bfrag(const uint32_t* packed, int fragIdx, uint32_t* u) {
  const uint32_t* pp = packed + (size_t)fragIdx * 256 + (threadIdx.x & 31) * 8;
#pragma unroll
  for (int v = 0; v < 8; ++v) u[v] = pp[v];
}
// K-split partial reduction into LDS f32 accumulator (ds_add_f32)
__device__ inline void acc_red(float* red, int S, int mt, int ntLocal, v8f c) {
  int lane = threadIdx.x & 31;
  int col  = ntLocal * 16 + (lane & 15);
  int half = lane >> 4;
#pragma unroll
  for (int v = 0; v < 8; ++v)
    atomicAdd(&red[(mt * 16 + v + 8 * half) * S + col], c[v]);
}

// ---------------- cross-WG barrier (cluster barrier + atomic fallback) --------
__device__ inline void grid_barrier(uint32_t* bar, int& phase) {
  __syncthreads();
  ++phase;
  if (threadIdx.x == 0) {
    __builtin_amdgcn_fence(__ATOMIC_RELEASE, "agent");
    __builtin_amdgcn_s_cluster_barrier();   // s_barrier_signal/-wait -3 (NOP outside cluster)
    __hip_atomic_fetch_add(bar, 1u, __ATOMIC_RELAXED, __HIP_MEMORY_SCOPE_AGENT);
    uint32_t tgt = (uint32_t)phase * NWG;
    while (__hip_atomic_load(bar, __ATOMIC_RELAXED, __HIP_MEMORY_SCOPE_AGENT) < tgt)
      __builtin_amdgcn_s_sleep(1);
    __builtin_amdgcn_fence(__ATOMIC_ACQUIRE, "agent");
  }
  __syncthreads();
}

// ---------------- init kernels ----------------
__global__ void zero_ws_kernel(uint32_t* dst, int n) {
  int i = blockIdx.x * blockDim.x + threadIdx.x;
  if (i < n) dst[i] = 0u;
}
// Pack W (logical K x N, f32) into WMMA B-fragment lane order (bf16 pairs)
__global__ void pack_kernel(uint32_t* dst, const float* src, int K, int N, int srcS,
                            int transp, int estride, int eoff) {
  int frag = blockIdx.x;
  int NT = (N + 15) >> 4;
  int kt = frag / NT, nt = frag % NT;
  int tid = threadIdx.x;
  int lane = tid >> 3, v = tid & 7;
  int col  = nt * 16 + (lane & 15);
  int half = (lane >> 4) & 1;
  int k0 = kt * 32 + ((v < 4) ? (8 * half + 2 * v) : (16 + 8 * half + 2 * (v - 4)));
  uint32_t lo = 0, hi = 0;
  if (col < N) {
    if (k0 < K) {
      size_t si = (size_t)(transp ? (col * srcS + k0) : (k0 * srcS + col)) * estride + eoff;
      lo = f2bf_u(src[si]);
    }
    if (k0 + 1 < K) {
      size_t si = (size_t)(transp ? (col * srcS + k0 + 1) : ((k0 + 1) * srcS + col)) * estride + eoff;
      hi = f2bf_u(src[si]);
    }
  }
  dst[(size_t)frag * 256 + tid] = lo | (hi << 16);
}

// ---------------- persistent decoder kernel ----------------
struct Params {
  const float *enc_z, *pre_b1, *pre_b2, *att_bs, *att_Wp, *att_bp;
  const float *lstm_b1, *lstm_b2, *stop_W, *stop_b;
  const float *cb0, *cb1, *cb2, *cb3, *cb4;
  float* out;
  uint8_t* ws;
};

__global__ __launch_bounds__(NTHR, 1) void decoder_kernel(Params p) {
  extern __shared__ uint8_t smem[];
  const int tid  = threadIdx.x;
  const int lane = tid & 31;
  const int wv   = tid >> 5;
  const int wg   = blockIdx.x;

  uint32_t* ldsWc   = (uint32_t*)(smem + L_WC);
  uint32_t* ldsWih1 = (uint32_t*)(smem + L_WIH1);
  uint32_t* ldsWih2 = (uint32_t*)(smem + L_WIH2);
  uint32_t* ldsC0   = (uint32_t*)(smem + L_C0);
  uint32_t* ldsC123 = (uint32_t*)(smem + L_C123);
  uint32_t* ldsC4   = (uint32_t*)(smem + L_C4);
  float*    stateF  = (float*)(smem + L_STATE);
  float*    muF     = (float*)(smem + L_MU);
  float*    wMix    = (float*)(smem + L_WMIX);
  float*    sMix    = (float*)(smem + L_SMIX);
  uint16_t* bufA    = (uint16_t*)(smem + L_PA);
  uint16_t* bufB    = (uint16_t*)(smem + L_PB);
  uint8_t*  reuse   = smem + L_REUSE;
  uint16_t* xA      = (uint16_t*)(smem + L_XA);
  uint16_t* befA    = (uint16_t*)(smem + L_BEFA);
  float*    befF    = (float*)(smem + L_BEFF);
  float*    red     = (float*)(smem + L_RED);

  uint32_t* bar    = (uint32_t*)(p.ws + OFF_BAR);
  float* afterG    = (float*)(p.ws + OFF_AFTER);
  uint16_t* ctxB   = (uint16_t*)(p.ws + OFF_CTX);   // bf16 ctx carry/exchange
  float* stateG    = (float*)(p.ws + OFF_STATE);
  float* g1G       = (float*)(p.ws + OFF_G1);
  float* g2G       = (float*)(p.ws + OFF_G2);
  uint16_t* xg0    = (uint16_t*)(p.ws + OFF_XP0);
  uint16_t* xg1    = (uint16_t*)(p.ws + OFF_XP1);
  const uint32_t* wPre1 = (const uint32_t*)(p.ws + OFF_WPRE1);
  const uint32_t* wPre2 = (const uint32_t*)(p.ws + OFF_WPRE2);
  const uint32_t* wWd   = (const uint32_t*)(p.ws + OFF_WD);

  const int r  = wg / 5;           // lstm gate region 0:i 1:g 2:o  (wg<15)
  const int tt = wg % 5;
  const int ntg1 = (r == 0) ? tt : (r == 1) ? 10 + tt : 15 + tt;  // ntile in Wih*T

  // ---- preload static weight slices into LDS via Tensor Data Mover (once) ----
  if (tid == 0) {
    if (wg < 8)   // Wc slice: 16 rows of 1 frag (512 elem), interleave stride 8 frags
      tdm_load_tile((uint32_t)L_WC, p.ws + OFF_WC + (size_t)wg * 1024, 512, 16, 8 * 512);
    if (wg < 15) {
      tdm_load_tile((uint32_t)L_WIH1, p.ws + OFF_WIH1 + (size_t)ntg1 * 1024, 512, 20, 20 * 512);
      tdm_load_tile((uint32_t)L_WIH2, p.ws + OFF_WIH2 + (size_t)ntg1 * 1024, 512, 3, 20 * 512);
    }
    // conv0: 3 rows of 2 contiguous frags (1024 elem), stride 32 frags
    tdm_load_tile((uint32_t)L_C0, p.ws + OFF_C0 + (size_t)wg * 2048, 1024, 3, 32 * 512);
    tdm_load_tile((uint32_t)(L_C123 + 0 * 32768), p.ws + OFF_C1 + (size_t)wg * 2048, 1024, 16, 32 * 512);
    tdm_load_tile((uint32_t)(L_C123 + 1 * 32768), p.ws + OFF_C2 + (size_t)wg * 2048, 1024, 16, 32 * 512);
    tdm_load_tile((uint32_t)(L_C123 + 2 * 32768), p.ws + OFF_C3 + (size_t)wg * 2048, 1024, 16, 32 * 512);
    if (wg < 5)
      tdm_load_tile((uint32_t)L_C4, p.ws + OFF_C4 + (size_t)wg * 1024, 512, 16, 5 * 512);
  }
  for (int i = tid; i < Bdim * KMIX; i += NTHR) muF[i] = 0.f;
  for (int i = tid; i < Bdim * 96; i += NTHR) bufB[i] = 0;
  if (tid == 0) tdm_wait();
  __syncthreads();

  int phase = 0;
  for (int l = 0; l < Ldim; ++l) {
    // ---- S0: stage after -> bufA (bf16 32x96, zero pad); ctx -> xA via TDM (wg<8) ----
    if (wg < 8 && tid == 0)
      tdm_load_tile((uint32_t)L_XA, ctxB, Bdim * Edim, 1, 0);  // 32KB contiguous bf16
    for (int i = tid; i < Bdim * 96; i += NTHR) {
      int b = i / 96, c = i % 96;
      bufA[i] = (c < Ddim) ? (uint16_t)f2bf_u(afterG[b * Ddim + c]) : (uint16_t)0;
    }
    if (wg < 8 && tid == 0) tdm_wait();
    __syncthreads();

    // ---- S1: prenet (replicated; B-frags streamed from L2-resident ws) ----
    for (int t5 = wv; t5 < 10; t5 += 8) {
      int mt = t5 & 1, nt = t5 >> 1;
      v8f c; for (int i = 0; i < 8; ++i) c[i] = 0.f;
      for (int kt = 0; kt < 3; ++kt) {
        uint32_t a8[8], b8[8];
        load_afrag(bufA, 96, mt, kt, a8);
        load_bfrag(wPre1, kt * 5 + nt, b8);
        c = wmma_acc(a8, b8, c);
      }
      int col = nt * 16 + (lane & 15), half = lane >> 4;
      float bias = p.pre_b1[col];
#pragma unroll
      for (int v = 0; v < 8; ++v) {
        float val = c[v] + bias; val = val > 0.f ? val : 0.f;
        bufB[(mt * 16 + v + 8 * half) * 96 + col] = (uint16_t)f2bf_u(val);
      }
    }
    __syncthreads();
    for (int t5 = wv; t5 < 10; t5 += 8) {
      int mt = t5 & 1, nt = t5 >> 1;
      v8f c; for (int i = 0; i < 8; ++i) c[i] = 0.f;
      for (int kt = 0; kt < 3; ++kt) {
        uint32_t a8[8], b8[8];
        load_afrag(bufB, 96, mt, kt, a8);
        load_bfrag(wPre2, kt * 5 + nt, b8);
        c = wmma_acc(a8, b8, c);
      }
      int col = nt * 16 + (lane & 15), half = lane >> 4;
      float bias = p.pre_b2[col];
#pragma unroll
      for (int v = 0; v < 8; ++v) {
        float val = c[v] + bias; val = val > 0.f ? val : 0.f;
        bufA[(mt * 16 + v + 8 * half) * 96 + col] = (uint16_t)f2bf_u(val);
      }
    }
    __syncthreads();

    // ---- S2: state (split: wg<8, one 16-col tile each; K split over waves) ----
    for (int i = tid; i < Bdim * 16; i += NTHR) red[i] = 0.f;
    __syncthreads();
    if (wg < 8) {
      int mt = wv >> 2, ks = wv & 3;
      v8f c; for (int i = 0; i < 8; ++i) c[i] = 0.f;
      if (ks < 3) {  // pre @ att_Wd
        uint32_t a8[8], b8[8];
        load_afrag(bufA, 96, mt, ks, a8);
        load_bfrag(wWd, ks * 8 + wg, b8);
        c = wmma_acc(a8, b8, c);
      }
      for (int kt = ks; kt < 16; kt += 4) {  // ctx @ att_Wc (Wc slice in LDS)
        uint32_t a8[8], b8[8];
        load_afrag(xA, Edim, mt, kt, a8);
        load_bfrag(ldsWc, kt, b8);
        c = wmma_acc(a8, b8, c);
      }
      acc_red(red, 16, mt, 0, c);
    }
    __syncthreads();
    if (wg < 8)
      for (int i = tid; i < Bdim * 16; i += NTHR) {
        int b = i >> 4, cc = i & 15, col = wg * 16 + cc;
        stateG[b * Adim + col] = tanhf(red[i] + p.att_bs[col]);
      }
    grid_barrier(bar, phase);  // B1

    // ---- S3: mixture params + alignment (replicated, transcendental VALU) ----
    for (int i = tid; i < Bdim * Adim; i += NTHR) stateF[i] = stateG[i];
    __syncthreads();
    float* pbuf = red;  // 32x15 fits
    for (int i = tid; i < Bdim * 15; i += NTHR) {
      int b = i / 15, o = i % 15;
      float acc = p.att_bp[o];
      for (int k = 0; k < Adim; ++k) acc += stateF[b * Adim + k] * p.att_Wp[k * 15 + o];
      pbuf[i] = acc;
    }
    __syncthreads();
    if (tid < Bdim) {
      int b = tid;
      float lw[KMIX], mx = -1e30f;
      for (int k = 0; k < KMIX; ++k) { lw[k] = pbuf[b * 15 + k]; mx = fmaxf(mx, lw[k]); }
      float sum = 0.f;
      for (int k = 0; k < KMIX; ++k) { lw[k] = __expf(lw[k] - mx); sum += lw[k]; }
      for (int k = 0; k < KMIX; ++k) {
        wMix[b * KMIX + k] = lw[k] / sum;
        muF[b * KMIX + k] += softplusf(pbuf[b * 15 + 5 + k]);
        sMix[b * KMIX + k] = softplusf(pbuf[b * 15 + 10 + k]) + 1e-4f;
      }
    }
    __syncthreads();
    float* aB = (float*)reuse;
    for (int i = tid; i < Bdim * Tdim; i += NTHR) {
      int b = i >> 8; float t = (float)(i & 255);
      float acc = 0.f;
      for (int k = 0; k < KMIX; ++k) {
        float mu = muF[b * KMIX + k], s = sMix[b * KMIX + k], w = wMix[b * KMIX + k];
        acc += w * (sigm((t + 0.5f - mu) / s) - sigm((t - 0.5f - mu) / s));
      }
      aB[i] = acc;
    }
    __syncthreads();

    // ---- S4: ctx slice (32 E-cols per WG), prefetch enc_z stream; bf16 out ----
    {
      int e0 = wg * 32;
      for (int i = tid; i < Bdim * 32; i += NTHR) {
        int b = i >> 5, e = e0 + (i & 31);
        const float* zp = p.enc_z + (size_t)b * Tdim * Edim + e;
        float acc = 0.f;
        for (int t = 0; t < Tdim; ++t) {
          if ((t & 31) == 0) __builtin_prefetch(zp + (size_t)(t + 32) * Edim, 0, 1);
          acc += aB[b * Tdim + t] * zp[(size_t)t * Edim];
        }
        ctxB[b * Edim + e] = (uint16_t)f2bf_u(acc);
      }
    }
    grid_barrier(bar, phase);  // B2

    // ---- S5: rnn_in stage + lstm1 (wg<15: one gate tile; K split over waves) ----
    uint16_t* rnnA = (uint16_t*)reuse;
    for (int i = tid; i < Bdim * EA; i += NTHR) {
      int b = i / EA, c = i % EA;
      rnnA[i] = (c < Edim) ? ctxB[b * Edim + c]
                           : (uint16_t)f2bf_u(stateF[b * Adim + (c - Edim)]);
    }
    for (int i = tid; i < Bdim * 16; i += NTHR) red[i] = 0.f;
    __syncthreads();
    if (wg < 15) {
      int mt = wv >> 2, ks = wv & 3;
      v8f c; for (int i = 0; i < 8; ++i) c[i] = 0.f;
      for (int kt = ks; kt < 20; kt += 4) {
        uint32_t a8[8], b8[8];
        load_afrag(rnnA, EA, mt, kt, a8);
        load_bfrag(ldsWih1, kt, b8);
        c = wmma_acc(a8, b8, c);
      }
      acc_red(red, 16, mt, 0, c);
    }
    __syncthreads();
    if (wg < 15) {
      int bofs = (r == 0 ? 0 : r == 1 ? 2 * Ddim : 3 * Ddim) + tt * 16;  // skip dead f gate
      int gofs = r * Ddim + tt * 16;
      for (int i = tid; i < Bdim * 16; i += NTHR) {
        int b = i >> 4, cc = i & 15;
        g1G[b * 240 + gofs + cc] = red[i] + p.lstm_b1[bofs + cc];
      }
    }
    grid_barrier(bar, phase);  // B3

    // ---- S6: h1 combine + lstm2 (wg<15) ----
    for (int i = tid; i < Bdim * 96; i += NTHR) {
      int b = i / 96, c = i % 96;
      uint16_t v16 = 0;
      if (c < Ddim) {
        float gi = g1G[b * 240 + c], gg = g1G[b * 240 + 80 + c], go = g1G[b * 240 + 160 + c];
        float cv = sigm(gi) * tanhf(gg);
        v16 = (uint16_t)f2bf_u(sigm(go) * tanhf(cv));
      }
      bufA[i] = v16;
    }
    __syncthreads();
    if (wg < 15 && wv < 2) {
      int mt = wv;
      v8f c; for (int i = 0; i < 8; ++i) c[i] = 0.f;
      for (int kt = 0; kt < 3; ++kt) {
        uint32_t a8[8], b8[8];
        load_afrag(bufA, 96, mt, kt, a8);
        load_bfrag(ldsWih2, kt, b8);
        c = wmma_acc(a8, b8, c);
      }
      int bofs = (r == 0 ? 0 : r == 1 ? 2 * Ddim : 3 * Ddim) + tt * 16;
      int gofs = r * Ddim + tt * 16;
      int cc = lane & 15, half = lane >> 4;
#pragma unroll
      for (int v = 0; v < 8; ++v)
        g2G[(mt * 16 + v + 8 * half) * 240 + gofs + cc] = c[v] + p.lstm_b2[bofs + cc];
    }
    grid_barrier(bar, phase);  // B4

    // ---- S7: before combine, outputs, postnet conv0 ----
    for (int i = tid; i < Bdim * 96; i += NTHR) {
      int b = i / 96, c = i % 96;
      uint16_t v16 = 0;
      if (c < Ddim) {
        float gi = g2G[b * 240 + c], gg = g2G[b * 240 + 80 + c], go = g2G[b * 240 + 160 + c];
        float cv = sigm(gi) * tanhf(gg);
        float h = sigm(go) * tanhf(cv);
        befF[b * Ddim + c] = h;
        v16 = (uint16_t)f2bf_u(h);
        if (wg == 0) p.out[BLD + (size_t)b * Ldim * Ddim + (size_t)l * Ddim + c] = h;
      }
      befA[i] = v16;
    }
    __syncthreads();
    if (wg == 0 && tid < Bdim) {
      float acc = p.stop_b[0];
      for (int d = 0; d < Ddim; ++d) acc += befF[tid * Ddim + d] * p.stop_W[d];
      p.out[2 * BLD + (size_t)tid * Ldim + l] = acc;
    }
    for (int i = tid; i < Bdim * 32; i += NTHR) red[i] = 0.f;
    __syncthreads();
    {
      int tile = wv >> 1, mt = tile >> 1, ntl = tile & 1, khalf = wv & 1;
      v8f c; for (int i = 0; i < 8; ++i) c[i] = 0.f;
      for (int kt = khalf; kt < 3; kt += 2) {
        uint32_t a8[8], b8[8];
        load_afrag(befA, 96, mt, kt, a8);
        load_bfrag(ldsC0, kt * 2 + ntl, b8);
        c = wmma_acc(a8, b8, c);
      }
      acc_red(red, 32, mt, ntl, c);
    }
    __syncthreads();
    for (int i = tid; i < Bdim * 32; i += NTHR) {
      int b = i >> 5, cc = i & 31, col = wg * 32 + cc;
      xg0[b * Cdim + col] = (uint16_t)f2bf_u(tanhf(red[i] + p.cb0[col]));
    }
    grid_barrier(bar, phase);  // B5

    // ---- S8: conv1..conv3 (weights LDS-resident; TDM stages exchange) ----
    for (int li = 0; li < 3; ++li) {
      const uint16_t* src = (li & 1) ? xg1 : xg0;
      uint16_t* dstx = (li & 1) ? xg0 : xg1;
      const float* cb = (li == 0) ? p.cb1 : (li == 1) ? p.cb2 : p.cb3;
      if (tid == 0) tdm_load_tile((uint32_t)L_XA, src, Bdim * Cdim, 1, 0);
      for (int i = tid; i < Bdim * 32; i += NTHR) red[i] = 0.f;
      if (tid == 0) tdm_wait();
      __syncthreads();
      {
        int tile = wv >> 1, mt = tile >> 1, ntl = tile & 1, khalf = wv & 1;
        v8f c; for (int i = 0; i < 8; ++i) c[i] = 0.f;
        const uint32_t* bw = ldsC123 + (size_t)li * 32 * 256;
        for (int kt = khalf * 8; kt < khalf * 8 + 8; ++kt) {
          uint32_t a8[8], b8[8];
          load_afrag(xA, Cdim, mt, kt, a8);
          load_bfrag(bw, kt * 2 + ntl, b8);
          c = wmma_acc(a8, b8, c);
        }
        acc_red(red, 32, mt, ntl, c);
      }
      __syncthreads();
      for (int i = tid; i < Bdim * 32; i += NTHR) {
        int b = i >> 5, cc = i & 31, col = wg * 32 + cc;
        dstx[b * Cdim + col] = (uint16_t)f2bf_u(tanhf(red[i] + cb[col]));
      }
      grid_barrier(bar, phase);  // B6/B7/B8
    }

    // ---- S9: conv4 (wg<5) + residual -> after carry + output ----
    if (wg < 5 && tid == 0) tdm_load_tile((uint32_t)L_XA, xg1, Bdim * Cdim, 1, 0);  // x3
    for (int i = tid; i < Bdim * 16; i += NTHR) red[i] = 0.f;
    if (wg < 5 && tid == 0) tdm_wait();
    __syncthreads();
    if (wg < 5) {
      int mt = wv >> 2, ks = wv & 3;
      v8f c; for (int i = 0; i < 8; ++i) c[i] = 0.f;
      for (int kt = ks; kt < 16; kt += 4) {
        uint32_t a8[8], b8[8];
        load_afrag(xA, Cdim, mt, kt, a8);
        load_bfrag(ldsC4, kt, b8);
        c = wmma_acc(a8, b8, c);
      }
      acc_red(red, 16, mt, 0, c);
    }
    __syncthreads();
    if (wg < 5)
      for (int i = tid; i < Bdim * 16; i += NTHR) {
        int b = i >> 4, cc = i & 15, col = wg * 16 + cc;
        float aft = befF[b * Ddim + col] + red[i] + p.cb4[col];
        afterG[b * Ddim + col] = aft;
        p.out[(size_t)b * Ldim * Ddim + (size_t)l * Ddim + col] = aft;
      }
    grid_barrier(bar, phase);  // B9 (protects after/ctx carries for next step)
  }
}

// ---------------- host launcher ----------------
extern "C" void kernel_launch(void* const* d_in, const int* in_sizes, int n_in,
                              void* d_out, int out_size, void* d_ws, size_t ws_size,
                              hipStream_t stream) {
  (void)in_sizes; (void)n_in; (void)out_size; (void)ws_size;
  uint8_t* ws = (uint8_t*)d_ws;

  const float* enc_z  = (const float*)d_in[0];
  const float* preW1  = (const float*)d_in[2];
  const float* preB1  = (const float*)d_in[3];
  const float* preW2  = (const float*)d_in[4];
  const float* preB2  = (const float*)d_in[5];
  const float* attWd  = (const float*)d_in[6];
  const float* attWc  = (const float*)d_in[7];
  const float* attBs  = (const float*)d_in[8];
  const float* attWp  = (const float*)d_in[9];
  const float* attBp  = (const float*)d_in[10];
  const float* wih1   = (const float*)d_in[11];
  const float* lb1    = (const float*)d_in[12];
  const float* wih2   = (const float*)d_in[13];
  const float* lb2    = (const float*)d_in[14];
  const float* stopW  = (const float*)d_in[15];
  const float* stopB  = (const float*)d_in[16];
  const float* cw0    = (const float*)d_in[17];
  const float* cb0    = (const float*)d_in[18];
  const float* cw1    = (const float*)d_in[19];
  const float* cb1    = (const float*)d_in[20];
  const float* cw2    = (const float*)d_in[21];
  const float* cb2    = (const float*)d_in[22];
  const float* cw3    = (const float*)d_in[23];
  const float* cb3    = (const float*)d_in[24];
  const float* cw4    = (const float*)d_in[25];
  const float* cb4    = (const float*)d_in[26];

  // zero barrier + carries (after f32, ctx bf16)
  zero_ws_kernel<<<(10816 + 255) / 256, 256, 0, stream>>>((uint32_t*)ws, 10816);

  // pack weights to WMMA B-fragment order (bf16) — L2-resident for the run
  pack_kernel<<<15, 256, 0, stream>>>((uint32_t*)(ws + OFF_WPRE1), preW1, 80, 80, 80, 0, 1, 0);
  pack_kernel<<<15, 256, 0, stream>>>((uint32_t*)(ws + OFF_WPRE2), preW2, 80, 80, 80, 0, 1, 0);
  pack_kernel<<<24, 256, 0, stream>>>((uint32_t*)(ws + OFF_WD),   attWd, 80, 128, 128, 0, 1, 0);
  pack_kernel<<<128, 256, 0, stream>>>((uint32_t*)(ws + OFF_WC),  attWc, 512, 128, 128, 0, 1, 0);
  pack_kernel<<<400, 256, 0, stream>>>((uint32_t*)(ws + OFF_WIH1), wih1, 640, 320, 640, 1, 1, 0);
  pack_kernel<<<60, 256, 0, stream>>>((uint32_t*)(ws + OFF_WIH2),  wih2, 80, 320, 80, 1, 1, 0);
  pack_kernel<<<96, 256, 0, stream>>>((uint32_t*)(ws + OFF_C0),    cw0, 80, 512, 80, 1, 5, 2);
  pack_kernel<<<512, 256, 0, stream>>>((uint32_t*)(ws + OFF_C1),   cw1, 512, 512, 512, 1, 5, 2);
  pack_kernel<<<512, 256, 0, stream>>>((uint32_t*)(ws + OFF_C2),   cw2, 512, 512, 512, 1, 5, 2);
  pack_kernel<<<512, 256, 0, stream>>>((uint32_t*)(ws + OFF_C3),   cw3, 512, 512, 512, 1, 5, 2);
  pack_kernel<<<80, 256, 0, stream>>>((uint32_t*)(ws + OFF_C4),    cw4, 512, 80, 512, 1, 5, 2);

  (void)hipFuncSetAttribute((const void*)decoder_kernel,
                            hipFuncAttributeMaxDynamicSharedMemorySize, SMEM_BYTES);

  Params P;
  P.enc_z = enc_z; P.pre_b1 = preB1; P.pre_b2 = preB2;
  P.att_bs = attBs; P.att_Wp = attWp; P.att_bp = attBp;
  P.lstm_b1 = lb1; P.lstm_b2 = lb2; P.stop_W = stopW; P.stop_b = stopB;
  P.cb0 = cb0; P.cb1 = cb1; P.cb2 = cb2; P.cb3 = cb3; P.cb4 = cb4;
  P.out = (float*)d_out; P.ws = ws;

  decoder_kernel<<<NWG, NTHR, SMEM_BYTES, stream>>>(P);
}
